// Model_35974646071616
// MI455X (gfx1250) — matrix operand order repaired and checked by reading, not compiled, not run on hardware
//
#include <hip/hip_runtime.h>
#include <hip/hip_bf16.h>
#include <math.h>

typedef __attribute__((ext_vector_type(16))) __bf16        v16bf;
typedef __attribute__((ext_vector_type(2)))  __bf16        v2bf;
typedef __attribute__((ext_vector_type(8)))  float         v8f;
typedef __attribute__((ext_vector_type(8)))  unsigned int  v8u;

// ---------------- model constants ----------------
constexpr int NB    = 32;     // batch
constexpr int LS    = 96;     // encoder seq len
constexpr int LDn   = 768;    // decoder seq len (LABEL+PRED)
constexpr int LABEL = 48;
constexpr int PRED  = 720;
constexpr int DM    = 512;    // d_model
constexpr int DFF   = 2048;
constexpr int CIN   = 7;
constexpr int COUT  = 7;
constexpr int NQm   = 64;     // kept q modes
constexpr int NKVm  = 48;     // kept k modes
constexpr int TK_E  = 13;     // int(3*ln(96))
constexpr int TK_D  = 19;     // int(3*ln(768))

static inline int cdiv(long a, long b) { return (int)((a + b - 1) / b); }

// pack two fp32 into one dword of bf16 pairs (lowers to v_cvt_pk_bf16_f32)
__device__ __forceinline__ unsigned pk2(float a, float b)
{
    v2bf p;
    p[0] = (__bf16)a; p[1] = (__bf16)b;
    return __builtin_bit_cast(unsigned, p);
}

// load 16 bf16 (two 16B groups) from LDS into a WMMA fragment
__device__ __forceinline__ v16bf ldfrag2(const unsigned* S, int off0, int off1)
{
    const uint4 q0 = *(const uint4*)(S + off0);
    const uint4 q1 = *(const uint4*)(S + off1);
    v8u t;
    t[0] = q0.x; t[1] = q0.y; t[2] = q0.z; t[3] = q0.w;
    t[4] = q1.x; t[5] = q1.y; t[6] = q1.z; t[7] = q1.w;
    return __builtin_bit_cast(v16bf, t);
}

// issue all global loads for one 64x32 A-slice and 64x32 B-slice (no waits here)
__device__ __forceinline__ void stage_global(const float* __restrict__ Ab,
                                             const float* __restrict__ Wb,
                                             int M, int N, int K,
                                             int tileM, int tileN, int k0, int tid,
                                             float4 va[4], float4 vb[4])
{
    #pragma unroll
    for (int it = 0; it < 4; ++it) {
        const int g = tid + it * 128;        // 0..511
        const int row = g >> 3, c4 = g & 7;
        int gra = tileM + row; if (gra > M - 1) gra = M - 1;
        int grb = tileN + row; if (grb > N - 1) grb = N - 1;
        va[it] = *(const float4*)(Ab + (long)gra * K + k0 + c4 * 4);
        vb[it] = *(const float4*)(Wb + (long)grb * K + k0 + c4 * 4);
    }
}

// convert staged registers to bf16 and store into an LDS buffer
__device__ __forceinline__ void stage_lds(unsigned* As, unsigned* Bs, int tid,
                                          const float4 va[4], const float4 vb[4])
{
    #pragma unroll
    for (int it = 0; it < 4; ++it) {
        const int g = tid + it * 128;
        const int row = g >> 3, c4 = g & 7;
        As[row * 16 + c4 * 2]     = pk2(va[it].x, va[it].y);
        As[row * 16 + c4 * 2 + 1] = pk2(va[it].z, va[it].w);
        Bs[row * 16 + c4 * 2]     = pk2(vb[it].x, vb[it].y);
        Bs[row * 16 + c4 * 2 + 1] = pk2(vb[it].z, vb[it].w);
    }
}

// =====================================================================
// WMMA bf16 GEMM:  C[m,n] = act( sum_k A[m,k]*W[n,k] + bias[n] ) + resid[m,n]
// A: (M,K) fp32 row-major, W: (N,K) fp32 row-major (i.e. x @ W^T).
// 128 threads (4 waves) per block -> 64x64 tile; each wave 32x32 via
// 2x2 WMMA subtiles. A/B K-slices double-buffered in LDS as bf16:
// global loads for tile k+1 are in flight while the 4 WMMAs consume
// tile k; one barrier per K-step. act: 0 = none, 1 = exact GELU.
// Partial M/N tiles: loads clamped, stores guarded (K % 32 == 0).
// =====================================================================
__global__ __launch_bounds__(128)
void wmma_gemm_bf16(const float* __restrict__ A, const float* __restrict__ W,
                    const float* __restrict__ bias, const float* __restrict__ resid,
                    float* __restrict__ C, int M, int N, int K,
                    long sA, long sW, long sC, int act)
{
    __shared__ uint4 AsV[2][256];    // 2 x (64 rows x 32 k bf16) = 8 KB
    __shared__ uint4 BsV[2][256];    // 8 KB

    const int bz = blockIdx.z;
    const float* Ab = A + (long)bz * sA;
    const float* Wb = W + (long)bz * sW;
    float*       Cb = C + (long)bz * sC;
    const float* Rb = resid ? (resid + (long)bz * sC) : nullptr;

    const int tileN = blockIdx.x * 64;
    const int tileM = blockIdx.y * 64;
    const int tid   = threadIdx.x;
    const int lane  = tid & 31;
    const int wid   = tid >> 5;
    const int wm    = (wid >> 1) * 32;       // wave row offset within tile
    const int wn    = (wid & 1) * 32;        // wave col offset within tile
    const int l15   = lane & 15;
    const bool hi   = lane >= 16;
    const int ka0u  = hi ? 4 : 0;            // A-frag uint offset (ka0/2)
    const int kb0u  = hi ? 4 : 0;            // B-frag uint offset (kb0/2)

    v8f acc00 = {0.f,0.f,0.f,0.f,0.f,0.f,0.f,0.f};
    v8f acc01 = acc00, acc10 = acc00, acc11 = acc00;

    const int prow_a = tileM + (tid & 63) > M - 1 ? M - 1 : tileM + (tid & 63);
    const int prow_b = tileN + (tid & 63) > N - 1 ? N - 1 : tileN + (tid & 63);

    // prologue: stage first K-slice into buffer 0
    float4 va[4], vb[4];
    stage_global(Ab, Wb, M, N, K, tileM, tileN, 0, tid, va, vb);
    stage_lds((unsigned*)AsV[0], (unsigned*)BsV[0], tid, va, vb);
    __syncthreads();

    const int ra0 = (wm + l15) * 16,      ra1 = (wm + 16 + l15) * 16;
    const int rb0 = (wn + l15) * 16,      rb1 = (wn + 16 + l15) * 16;

    for (int k0 = 0; k0 < K; k0 += 32) {
        const int cb = (k0 >> 5) & 1;
        const unsigned* Asc = (const unsigned*)AsV[cb];
        const unsigned* Bsc = (const unsigned*)BsV[cb];
        const bool more = (k0 + 32 < K);

        // kick off next slice's global loads (latency hidden behind WMMAs)
        if (more) {
            stage_global(Ab, Wb, M, N, K, tileM, tileN, k0 + 32, tid, va, vb);
            if (k0 + 64 < K) {
                __builtin_prefetch(Ab + (long)prow_a * K + k0 + 64, 0, 1);
                __builtin_prefetch(Wb + (long)prow_b * K + k0 + 64, 0, 1);
            }
        }

        // ---- fragments (ISA 7.12.2 layouts) + 2x2 WMMA on current slice ----
        const v16bf a0 = ldfrag2(Asc, ra0 + ka0u, ra0 + 8 + ka0u);
        const v16bf a1 = ldfrag2(Asc, ra1 + ka0u, ra1 + 8 + ka0u);
        const v16bf b0 = ldfrag2(Bsc, rb0 + kb0u, rb0 + kb0u + 8);
        const v16bf b1 = ldfrag2(Bsc, rb1 + kb0u, rb1 + kb0u + 8);

        acc00 = __builtin_amdgcn_wmma_f32_16x16x32_bf16(false, a0, false, b0, (short)0, acc00, false, false);
        acc01 = __builtin_amdgcn_wmma_f32_16x16x32_bf16(false, a0, false, b1, (short)0, acc01, false, false);
        acc10 = __builtin_amdgcn_wmma_f32_16x16x32_bf16(false, a1, false, b0, (short)0, acc10, false, false);
        acc11 = __builtin_amdgcn_wmma_f32_16x16x32_bf16(false, a1, false, b1, (short)0, acc11, false, false);

        // convert + publish next slice into the other buffer
        if (more)
            stage_lds((unsigned*)AsV[cb ^ 1], (unsigned*)BsV[cb ^ 1], tid, va, vb);
        __syncthreads();
    }

    // ---- epilogue: bias, activation, residual, guarded store ----
    #pragma unroll
    for (int i = 0; i < 2; ++i) {
        #pragma unroll
        for (int j = 0; j < 2; ++j) {
            const v8f acc = (i == 0) ? (j == 0 ? acc00 : acc01)
                                     : (j == 0 ? acc10 : acc11);
            const int col = tileN + wn + j * 16 + l15;
            if (col >= N) continue;
            const float bv = bias ? bias[col] : 0.0f;
            #pragma unroll
            for (int r = 0; r < 8; ++r) {
                const int row = tileM + wm + i * 16 + r + (hi ? 8 : 0);
                if (row >= M) continue;
                float v = acc[r] + bv;
                if (act == 1) v = 0.5f * v * (1.0f + erff(v * 0.70710678118f));
                if (Rb) v += Rb[(long)row * N + col];
                Cb[(long)row * N + col] = v;
            }
        }
    }
}

// =====================================================================
// series_decomp: 25-tap replicate-padded moving mean.
// seasonal = x - mean (must differ from x); trend optional (+accumulate).
// =====================================================================
__global__ void k_decomp(const float* __restrict__ x, float* __restrict__ seasonal,
                         float* __restrict__ trend, int B, int L, int C, int acc)
{
    long i = (long)blockIdx.x * blockDim.x + threadIdx.x;
    long total = (long)B * L * C;
    if (i >= total) return;
    int c = (int)(i % C); long t = i / C; int l = (int)(t % L); int b = (int)(t / L);
    const float* xb = x + ((long)b * L) * C + c;
    float s = 0.f;
    #pragma unroll 5
    for (int j = -12; j <= 12; ++j) {
        int ll = l + j; ll = ll < 0 ? 0 : (ll >= L ? L - 1 : ll);
        s += xb[(long)ll * C];
    }
    float m = s * (1.0f / 25.0f);
    if (trend) { if (acc) trend[i] += m; else trend[i] = m; }
    seasonal[i] = x[i] - m;
}

__global__ void k_mean_seq(const float* __restrict__ x, float* __restrict__ mean)
{
    int i = blockIdx.x * blockDim.x + threadIdx.x;
    if (i >= NB * CIN) return;
    int c = i % CIN, b = i / CIN;
    float s = 0.f;
    for (int l = 0; l < LS; ++l) s += x[((long)b * LS + l) * CIN + c];
    mean[i] = s / (float)LS;
}

__global__ void k_trend_init(const float* __restrict__ trend_x, const float* __restrict__ mean,
                             float* __restrict__ out)
{
    long i = (long)blockIdx.x * blockDim.x + threadIdx.x;
    if (i >= (long)NB * LDn * COUT) return;
    int c = (int)(i % COUT); long t = i / COUT; int l = (int)(t % LDn); int b = (int)(t / LDn);
    out[i] = (l < LABEL) ? trend_x[((long)b * LS + (LS - LABEL + l)) * CIN + c]
                         : mean[b * CIN + c];
}

__global__ void k_season_init(const float* __restrict__ sx, float* __restrict__ out)
{
    long i = (long)blockIdx.x * blockDim.x + threadIdx.x;
    if (i >= (long)NB * LDn * CIN) return;
    int c = (int)(i % CIN); long t = i / CIN; int l = (int)(t % LDn); int b = (int)(t / LDn);
    out[i] = (l < LABEL) ? sx[((long)b * LS + (LS - LABEL + l)) * CIN + c] : 0.0f;
}

// circular conv, kernel 3:  out[b,l,o] (+)= sum_{j,c} x[b,(l+j-1)%L,c]*w[o,c,j]
__global__ void k_conv3(const float* __restrict__ x, const float* __restrict__ w,
                        float* __restrict__ out, int B, int L, int Cin, int Cout, int acc)
{
    long i = (long)blockIdx.x * blockDim.x + threadIdx.x;
    long total = (long)B * L * Cout;
    if (i >= total) return;
    int o = (int)(i % Cout); long t = i / Cout; int l = (int)(t % L); int b = (int)(t / L);
    const float* xb = x + (long)b * L * Cin;
    float s = 0.f;
    for (int j = 0; j < 3; ++j) {
        int ll = l + j - 1; if (ll < 0) ll += L; if (ll >= L) ll -= L;
        const float* xr = xb + (long)ll * Cin;
        const float* wr = w + ((long)o * Cin) * 3 + j;
        for (int c = 0; c < Cin; ++c) s += xr[c] * wr[(long)c * 3];
    }
    if (acc) out[i] += s; else out[i] = s;
}

// mean_corr[b,tau] = (1/512) * sum_t G[b,(t+tau)%L, t]
__global__ void k_diagcorr(const float* __restrict__ G, float* __restrict__ mc, int B, int L)
{
    int i = blockIdx.x * blockDim.x + threadIdx.x;
    if (i >= B * L) return;
    int tau = i % L, b = i / L;
    const float* Gb = G + (long)b * L * L;
    float s = 0.f;
    for (int t = 0; t < L; ++t) {
        int r = t + tau; if (r >= L) r -= L;
        s += Gb[(long)r * L + t];
    }
    mc[i] = s * (1.0f / 512.0f);
}

// one block per batch: iterative top-k + softmax of the selected weights
__global__ void k_topk(const float* __restrict__ mc, float* __restrict__ wsm,
                       int* __restrict__ delays, int L, int topk)
{
    __shared__ float vals[768];
    __shared__ float rv[256];
    __shared__ int   ri[256];
    __shared__ float tw[32];
    const int b = blockIdx.x, tid = threadIdx.x, nt = blockDim.x;
    for (int i = tid; i < L; i += nt) vals[i] = mc[b * L + i];
    __syncthreads();
    for (int kk = 0; kk < topk; ++kk) {
        float best = -1e30f; int bi = 0;
        for (int i = tid; i < L; i += nt)
            if (vals[i] > best) { best = vals[i]; bi = i; }
        rv[tid] = best; ri[tid] = bi;
        __syncthreads();
        for (int s = nt >> 1; s > 0; s >>= 1) {
            if (tid < s && rv[tid + s] > rv[tid]) { rv[tid] = rv[tid + s]; ri[tid] = ri[tid + s]; }
            __syncthreads();
        }
        if (tid == 0) { tw[kk] = rv[0]; delays[b * 32 + kk] = ri[0]; vals[ri[0]] = -1e30f; }
        __syncthreads();
    }
    if (tid == 0) {
        float mx = -1e30f;
        for (int i = 0; i < topk; ++i) mx = fmaxf(mx, tw[i]);
        float se = 0.f;
        for (int i = 0; i < topk; ++i) { tw[i] = expf(tw[i] - mx); se += tw[i]; }
        for (int i = 0; i < topk; ++i) wsm[b * 32 + i] = tw[i] / se;
    }
}

// out[b,l,c] = sum_i wsm[b,i] * v[b,(l+delay[b,i])%L, c]
__global__ void k_agg(const float* __restrict__ v, const float* __restrict__ wsm,
                      const int* __restrict__ delays, float* __restrict__ out,
                      int B, int L, int C, int topk)
{
    long i = (long)blockIdx.x * blockDim.x + threadIdx.x;
    long total = (long)B * L * C;
    if (i >= total) return;
    int c = (int)(i % C); long t = i / C; int l = (int)(t % L); int b = (int)(t / L);
    const float* vb = v + (long)b * L * C + c;
    float s = 0.f;
    for (int kk = 0; kk < topk; ++kk) {
        int ll = l + delays[b * 32 + kk]; if (ll >= L) ll -= L;
        s += wsm[b * 32 + kk] * vb[(long)ll * C];
    }
    out[i] = s;
}

// per-row layernorm over D=512, one wave per row
__global__ __launch_bounds__(32)
void k_ln_rows(const float* __restrict__ x, const float* __restrict__ w,
               const float* __restrict__ bb, float* __restrict__ y)
{
    const long r = blockIdx.x;
    const int lane = threadIdx.x;
    const float* xr = x + r * DM;
    float s = 0.f;
    for (int i = lane; i < DM; i += 32) s += xr[i];
    #pragma unroll
    for (int o = 16; o > 0; o >>= 1) s += __shfl_xor(s, o, 32);
    const float mu = s / (float)DM;
    float v = 0.f;
    for (int i = lane; i < DM; i += 32) { float d = xr[i] - mu; v += d * d; }
    #pragma unroll
    for (int o = 16; o > 0; o >>= 1) v += __shfl_xor(v, o, 32);
    const float inv = rsqrtf(v / (float)DM + 1e-5f);
    float* yr = y + r * DM;
    for (int i = lane; i < DM; i += 32) yr[i] = (xr[i] - mu) * inv * w[i] + bb[i];
}

// subtract per-(b,channel) mean over the time axis (my_layernorm tail)
__global__ void k_sub_colmean(float* __restrict__ x, int B, int L, int C)
{
    int i = blockIdx.x * blockDim.x + threadIdx.x;
    if (i >= B * C) return;
    int c = i % C, b = i / C;
    float* xb = x + (long)b * L * C + c;
    float s = 0.f;
    for (int l = 0; l < L; ++l) s += xb[(long)l * C];
    s /= (float)L;
    for (int l = 0; l < L; ++l) xb[(long)l * C] -= s;
}

// mode-truncated DFT:  X[b,h,e,m] = sum_t x[b,t,(h,e)] * e^{-2pi i m t / L}
__global__ void k_rfft(const float* __restrict__ x, float* __restrict__ X,
                       int B, int L, int NM)
{
    long i = (long)blockIdx.x * blockDim.x + threadIdx.x;
    long total = (long)B * DM * NM;
    if (i >= total) return;
    int m = (int)(i % NM); long t = i / NM; int ch = (int)(t % DM); int b = (int)(t / DM);
    const float* xb = x + (long)b * L * DM + ch;
    const float w0 = -6.2831853071795864f * (float)m / (float)L;
    float re = 0.f, im = 0.f;
    for (int tt = 0; tt < L; ++tt) {
        float sn, cs; __sincosf(w0 * (float)tt, &sn, &cs);
        float v = xb[(long)tt * DM];
        re += v * cs; im += v * sn;
    }
    X[2 * i] = re; X[2 * i + 1] = im;
}

// xqk[b,h,x,y] = ctanh( sum_e xq[b,h,e,x] * xk[b,h,e,y] )
__global__ void k_xqk(const float* __restrict__ xq, const float* __restrict__ xk,
                      float* __restrict__ out)
{
    long i = (long)blockIdx.x * blockDim.x + threadIdx.x;
    long total = (long)NB * 8 * NQm * NKVm;
    if (i >= total) return;
    int y = (int)(i % NKVm); long t = i / NKVm; int xm = (int)(t % NQm);
    t /= NQm; int h = (int)(t % 8); int b = (int)(t / 8);
    const float* q = xq + (((long)(b * 8 + h) * 64) * NQm + xm) * 2;
    const float* k = xk + (((long)(b * 8 + h) * 64) * NKVm + y) * 2;
    float re = 0.f, im = 0.f;
    for (int e = 0; e < 64; ++e) {
        float qr = q[(long)e * NQm * 2], qi = q[(long)e * NQm * 2 + 1];
        float kr = k[(long)e * NKVm * 2], ki = k[(long)e * NKVm * 2 + 1];
        re += qr * kr - qi * ki;
        im += qr * ki + qi * kr;
    }
    // complex tanh(a+ib) = (ta(1+tb^2) + i tb(1-ta^2)) / (1 + ta^2 tb^2)
    float ta = tanhf(re), tb = tanf(im);
    float den = 1.0f + ta * ta * tb * tb;
    out[2 * i]     = ta * (1.0f + tb * tb) / den;
    out[2 * i + 1] = tb * (1.0f - ta * ta) / den;
}

// xqkv[b,h,e,x] = sum_y xqk[b,h,x,y] * xk[b,h,e,y]
__global__ void k_xqkv(const float* __restrict__ xqk, const float* __restrict__ xk,
                       float* __restrict__ out)
{
    long i = (long)blockIdx.x * blockDim.x + threadIdx.x;
    long total = (long)NB * 8 * 64 * NQm;
    if (i >= total) return;
    int xm = (int)(i % NQm); long t = i / NQm; int e = (int)(t % 64);
    t /= 64; int h = (int)(t % 8); int b = (int)(t / 8);
    const float* A = xqk + ((long)(b * 8 + h) * NQm + xm) * NKVm * 2;
    const float* K = xk + (((long)(b * 8 + h) * 64 + e) * NKVm) * 2;
    float re = 0.f, im = 0.f;
    for (int y = 0; y < NKVm; ++y) {
        float ar = A[2 * y], ai = A[2 * y + 1];
        float kr = K[2 * y], ki = K[2 * y + 1];
        re += ar * kr - ai * ki;
        im += ar * ki + ai * kr;
    }
    out[2 * i] = re; out[2 * i + 1] = im;
}

// xqkvw[b,h,o,x] = sum_e xqkv[b,h,e,x] * (wr + i*wi)[h,e,o,x]
__global__ void k_xqkvw(const float* __restrict__ xqkv, const float* __restrict__ wr,
                        const float* __restrict__ wi, float* __restrict__ out)
{
    long i = (long)blockIdx.x * blockDim.x + threadIdx.x;
    long total = (long)NB * 8 * 64 * NQm;
    if (i >= total) return;
    int xm = (int)(i % NQm); long t = i / NQm; int o = (int)(t % 64);
    t /= 64; int h = (int)(t % 8); int b = (int)(t / 8);
    float re = 0.f, im = 0.f;
    for (int e = 0; e < 64; ++e) {
        long vi = (((long)(b * 8 + h) * 64 + e) * NQm + xm) * 2;
        long wiidx = ((long)(h * 64 + e) * 64 + o) * NQm + xm;
        float vr = xqkv[vi], vii = xqkv[vi + 1];
        float r = wr[wiidx], ii = wi[wiidx];
        re += vr * r - vii * ii;
        im += vr * ii + vii * r;
    }
    out[2 * i] = re; out[2 * i + 1] = im;
}

// iDFT with only NQm low modes, scaled by 1/(L * D^2); out layout (b,l,h*64+o)
__global__ void k_irfft_cross(const float* __restrict__ X, float* __restrict__ out)
{
    long i = (long)blockIdx.x * blockDim.x + threadIdx.x;
    long total = (long)NB * LDn * DM;
    if (i >= total) return;
    int ch = (int)(i % DM); long t = i / DM; int l = (int)(t % LDn); int b = (int)(t / LDn);
    int h = ch >> 6, o = ch & 63;
    const float* Xp = X + (((long)(b * 8 + h) * 64 + o) * NQm) * 2;
    const float w0 = 6.2831853071795864f * (float)l / (float)LDn;
    float s = Xp[0];               // m = 0 (real part)
    for (int m = 1; m < NQm; ++m) {
        float sn, cs; __sincosf(w0 * (float)m, &sn, &cs);
        s += 2.0f * (Xp[2 * m] * cs - Xp[2 * m + 1] * sn);
    }
    out[i] = s * (1.0f / (float)LDn) * (1.0f / (512.0f * 512.0f));
}

// out[b, l-48, o] = trend[b,l,o] + x[b,l,:] @ w[o,:] + bias[o], l in [48,768)
__global__ void k_final(const float* __restrict__ x, const float* __restrict__ w,
                        const float* __restrict__ bias, const float* __restrict__ trend,
                        float* __restrict__ out)
{
    long i = (long)blockIdx.x * blockDim.x + threadIdx.x;
    long total = (long)NB * PRED * COUT;
    if (i >= total) return;
    int o = (int)(i % COUT); long t = i / COUT; int lp = (int)(t % PRED); int b = (int)(t / PRED);
    int l = lp + LABEL;
    const float* xr = x + ((long)b * LDn + l) * DM;
    const float* wr = w + (long)o * DM;
    float s = bias[o];
    for (int c = 0; c < DM; ++c) s += xr[c] * wr[c];
    out[i] = trend[((long)b * LDn + l) * COUT + o] + s;
}

// =====================================================================
// Host orchestration
// =====================================================================
extern "C" void kernel_launch(void* const* d_in, const int* in_sizes, int n_in,
                              void* d_out, int out_size, void* d_ws, size_t ws_size,
                              hipStream_t stream)
{
    (void)in_sizes; (void)n_in; (void)out_size; (void)ws_size;
    const float* x_enc     = (const float*)d_in[0];
    const float* enc_tok_w = (const float*)d_in[1];
    const float* dec_tok_w = (const float*)d_in[2];
    const float* enc_nw    = (const float*)d_in[23];
    const float* enc_nb    = (const float*)d_in[24];
    const float* dec_nw    = (const float*)d_in[46];
    const float* dec_nb    = (const float*)d_in[47];
    const float* dproj_w   = (const float*)d_in[48];
    const float* dproj_b   = (const float*)d_in[49];
    float* out = (float*)d_out;

    // -------- workspace carving --------
    char* ws = (char*)d_ws;
    size_t cur = 0;
    auto alloc = [&](size_t nfl) {
        float* p = (float*)(ws + cur);
        cur += ((nfl * 4 + 255) / 256) * 256;
        return p;
    };
    const long ENC_SZ = (long)NB * LS * DM;     // 1.57M fl
    const long DEC_SZ = (long)NB * LDn * DM;    // 12.58M fl
    const long G_SZ   = (long)NB * LDn * LDn;   // 18.87M fl

    float* E0 = alloc(ENC_SZ);
    float* E1 = alloc(ENC_SZ);
    float* E2 = alloc(ENC_SZ);
    float* E3 = alloc(ENC_SZ);
    float* D0 = alloc(DEC_SZ);
    float* D1 = alloc(DEC_SZ);
    float* D2 = alloc(DEC_SZ);
    float* D3 = alloc(DEC_SZ);
    float* Dt = alloc(DEC_SZ);                  // accumulated d_model trends
    float* GB = alloc(G_SZ);                    // Gram / FFN-hidden / cross scratch
    float* mc = alloc((long)NB * LDn);
    float* wsmb = alloc(NB * 32);
    int*   delays = (int*)alloc(NB * 32);
    float* meanb = alloc(NB * CIN);
    float* tx = alloc((long)NB * LS * CIN);
    float* sx = alloc((long)NB * LS * CIN);
    float* TTOT  = alloc((long)NB * LDn * COUT);
    float* SINIT = alloc((long)NB * LDn * CIN);
    // cross-attention complex buffers live inside GB (free during cross phase)
    float* XQFT  = GB;
    float* XKFT  = XQFT + (long)NB * 8 * 64 * NQm * 2;
    float* XQK   = XKFT + (long)NB * 8 * 64 * NKVm * 2;
    float* XQKV  = XQK  + (long)NB * 8 * NQm * NKVm * 2;
    float* XQKVW = XQKV + (long)NB * 8 * 64 * NQm * 2;

    auto gemm = [&](const float* A, const float* W, const float* bias,
                    const float* resid, float* C, int M, int N, int K,
                    int act, int Z, long sA, long sW, long sC) {
        dim3 g((unsigned)cdiv(N, 64), (unsigned)cdiv(M, 64), (unsigned)Z);
        wmma_gemm_bf16<<<g, dim3(128), 0, stream>>>(A, W, bias, resid, C,
                                                    M, N, K, sA, sW, sC, act);
    };

    // -------- preprocessing --------
    k_mean_seq<<<cdiv(NB * CIN, 64), 64, 0, stream>>>(x_enc, meanb);
    k_decomp<<<cdiv((long)NB * LS * CIN, 256), 256, 0, stream>>>(x_enc, sx, tx, NB, LS, CIN, 0);
    k_trend_init<<<cdiv((long)NB * LDn * COUT, 256), 256, 0, stream>>>(tx, meanb, TTOT);
    k_season_init<<<cdiv((long)NB * LDn * CIN, 256), 256, 0, stream>>>(sx, SINIT);
    k_conv3<<<cdiv((long)NB * LS * DM, 256), 256, 0, stream>>>(x_enc, enc_tok_w, E0, NB, LS, CIN, DM, 0);

    // -------- encoder (2 layers), current x = E0 --------
    for (int l = 0; l < 2; ++l) {
        const int bs = 3 + 10 * l;
        const float* wq = (const float*)d_in[bs + 0]; const float* bq = (const float*)d_in[bs + 1];
        const float* wk = (const float*)d_in[bs + 2]; const float* bk = (const float*)d_in[bs + 3];
        const float* wv = (const float*)d_in[bs + 4]; const float* bv = (const float*)d_in[bs + 5];
        const float* wo = (const float*)d_in[bs + 6]; const float* bo = (const float*)d_in[bs + 7];
        const float* c1 = (const float*)d_in[bs + 8]; const float* c2 = (const float*)d_in[bs + 9];

        gemm(E0, wq, bq, nullptr, E1, NB * LS, DM, DM, 0, 1, 0, 0, 0);
        gemm(E0, wk, bk, nullptr, E2, NB * LS, DM, DM, 0, 1, 0, 0, 0);
        gemm(E0, wv, bv, nullptr, E3, NB * LS, DM, DM, 0, 1, 0, 0, 0);
        // Gram G = Q K^T per batch (circular autocorrelation via diagonals)
        gemm(E1, E2, nullptr, nullptr, GB, LS, LS, DM, 0, NB,
             (long)LS * DM, (long)LS * DM, (long)LS * LS);
        k_diagcorr<<<cdiv(NB * LS, 256), 256, 0, stream>>>(GB, mc, NB, LS);
        k_topk<<<NB, 256, 0, stream>>>(mc, wsmb, delays, LS, TK_E);
        k_agg<<<cdiv((long)NB * LS * DM, 256), 256, 0, stream>>>(E3, wsmb, delays, E1, NB, LS, DM, TK_E);
        gemm(E1, wo, bo, E0, E2, NB * LS, DM, DM, 0, 1, 0, 0, 0);     // x' = x + attn
        k_decomp<<<cdiv(ENC_SZ, 256), 256, 0, stream>>>(E2, E0, nullptr, NB, LS, DM, 0);
        // FFN (hidden reuses GB)
        gemm(E0, c1, nullptr, nullptr, GB, NB * LS, DFF, DM, 1, 1, 0, 0, 0);
        gemm(GB, c2, nullptr, E0, E1, NB * LS, DM, DFF, 0, 1, 0, 0, 0);
        k_decomp<<<cdiv(ENC_SZ, 256), 256, 0, stream>>>(E1, E0, nullptr, NB, LS, DM, 0);
    }
    k_ln_rows<<<NB * LS, 32, 0, stream>>>(E0, enc_nw, enc_nb, E1);
    k_sub_colmean<<<cdiv(NB * DM, 256), 256, 0, stream>>>(E1, NB, LS, DM);
    float* enc_out = E1;

    // -------- decoder embed --------
    k_conv3<<<cdiv(DEC_SZ, 256), 256, 0, stream>>>(SINIT, dec_tok_w, D0, NB, LDn, CIN, DM, 0);

    // -------- decoder layer (1 layer, base idx 25) --------
    {
        const int bs = 25;
        const float* swq = (const float*)d_in[bs + 0]; const float* sbq = (const float*)d_in[bs + 1];
        const float* swk = (const float*)d_in[bs + 2]; const float* sbk = (const float*)d_in[bs + 3];
        const float* swv = (const float*)d_in[bs + 4]; const float* sbv = (const float*)d_in[bs + 5];
        const float* swo = (const float*)d_in[bs + 6]; const float* sbo = (const float*)d_in[bs + 7];
        const float* cwq = (const float*)d_in[33]; const float* cbq = (const float*)d_in[34];
        const float* cwk = (const float*)d_in[35]; const float* cbk = (const float*)d_in[36];
        const float* cwo = (const float*)d_in[39]; const float* cbo = (const float*)d_in[40];
        const float* cwr = (const float*)d_in[41]; const float* cwi = (const float*)d_in[42];
        const float* c1  = (const float*)d_in[43]; const float* c2  = (const float*)d_in[44];
        const float* tpw = (const float*)d_in[45];

        // --- self autocorrelation attention ---
        gemm(D0, swq, sbq, nullptr, D1, NB * LDn, DM, DM, 0, 1, 0, 0, 0);
        gemm(D0, swk, sbk, nullptr, D2, NB * LDn, DM, DM, 0, 1, 0, 0, 0);
        gemm(D0, swv, sbv, nullptr, D3, NB * LDn, DM, DM, 0, 1, 0, 0, 0);
        gemm(D1, D2, nullptr, nullptr, GB, LDn, LDn, DM, 0, NB,
             (long)LDn * DM, (long)LDn * DM, (long)LDn * LDn);
        k_diagcorr<<<cdiv(NB * LDn, 256), 256, 0, stream>>>(GB, mc, NB, LDn);
        k_topk<<<NB, 256, 0, stream>>>(mc, wsmb, delays, LDn, TK_D);
        k_agg<<<cdiv(DEC_SZ, 256), 256, 0, stream>>>(D3, wsmb, delays, D2, NB, LDn, DM, TK_D);
        gemm(D2, swo, sbo, D0, D1, NB * LDn, DM, DM, 0, 1, 0, 0, 0);  // x' = x + attn
        k_decomp<<<cdiv(DEC_SZ, 256), 256, 0, stream>>>(D1, D2, Dt, NB, LDn, DM, 0);  // t1

        // --- spectral cross attention (x = D2, cross input = enc_out) ---
        gemm(D2, cwq, cbq, nullptr, D1, NB * LDn, DM, DM, 0, 1, 0, 0, 0);
        gemm(enc_out, cwk, cbk, nullptr, E2, NB * LS, DM, DM, 0, 1, 0, 0, 0);
        k_rfft<<<cdiv((long)NB * DM * NQm, 256), 256, 0, stream>>>(D1, XQFT, NB, LDn, NQm);
        k_rfft<<<cdiv((long)NB * DM * NKVm, 256), 256, 0, stream>>>(E2, XKFT, NB, LS, NKVm);
        k_xqk<<<cdiv((long)NB * 8 * NQm * NKVm, 256), 256, 0, stream>>>(XQFT, XKFT, XQK);
        k_xqkv<<<cdiv((long)NB * 8 * 64 * NQm, 256), 256, 0, stream>>>(XQK, XKFT, XQKV);
        k_xqkvw<<<cdiv((long)NB * 8 * 64 * NQm, 256), 256, 0, stream>>>(XQKV, cwr, cwi, XQKVW);
        k_irfft_cross<<<cdiv(DEC_SZ, 256), 256, 0, stream>>>(XQKVW, D3);
        gemm(D3, cwo, cbo, D2, D1, NB * LDn, DM, DM, 0, 1, 0, 0, 0);  // x' = x + cross
        k_decomp<<<cdiv(DEC_SZ, 256), 256, 0, stream>>>(D1, D3, Dt, NB, LDn, DM, 1);  // t2

        // --- FFN (x = D3), hidden chunked through GB ---
        const int CH = 8192;                     // rows per chunk (3 chunks)
        for (int co = 0; co < NB * LDn / CH; ++co) {
            const float* xa = D3 + (long)co * CH * DM;
            gemm(xa, c1, nullptr, nullptr, GB, CH, DFF, DM, 1, 1, 0, 0, 0);
            gemm(GB, c2, nullptr, xa, D1 + (long)co * CH * DM, CH, DM, DFF, 0, 1, 0, 0, 0);
        }
        k_decomp<<<cdiv(DEC_SZ, 256), 256, 0, stream>>>(D1, D2, Dt, NB, LDn, DM, 1);  // t3

        // residual trend projection: TTOT += conv3(t1+t2+t3)
        k_conv3<<<cdiv((long)NB * LDn * COUT, 256), 256, 0, stream>>>(Dt, tpw, TTOT, NB, LDn, DM, COUT, 1);
    }

    // -------- decoder norm + final projection --------
    k_ln_rows<<<NB * LDn, 32, 0, stream>>>(D2, dec_nw, dec_nb, D1);
    k_sub_colmean<<<cdiv(NB * DM, 256), 256, 0, stream>>>(D1, NB, LDn, DM);
    k_final<<<cdiv((long)NB * PRED * COUT, 256), 256, 0, stream>>>(D1, dproj_w, dproj_b, TTOT, out);
}
